// MDMAttention_88940182765732
// MI455X (gfx1250) — matrix operand order repaired and checked by reading, not compiled, not run on hardware
//
#include <hip/hip_runtime.h>
#include <hip/hip_bf16.h>
#include <cmath>

typedef _Float16 f16;
typedef __attribute__((ext_vector_type(16))) _Float16 v16h;
typedef __attribute__((ext_vector_type(8)))  _Float16 v8h;
typedef __attribute__((ext_vector_type(4)))  _Float16 v4h;
typedef __attribute__((ext_vector_type(8)))  float    v8f;

#define N_EMBD  2048
#define T_SEQ   2048
#define BATCH   2
#define NH      32
#define NG      8
#define HD      64
#define QPK     4
#define QKV_DIM 3072   // (QPK+2)*NG*HD

// 1/sqrt(HD) * log2(e): folded into Q so attention softmax can use exp2
#define QSCALE  0.1803368801111204f

// ---------------------------------------------------------------------------
// WMMA helpers (CDNA5 wave32, V_WMMA_F32_16X16X32_F16)
// ---------------------------------------------------------------------------
static __device__ __forceinline__ v8f wmma_f16(v16h a, v16h b, v8f c) {
    return __builtin_amdgcn_wmma_f32_16x16x32_f16(false, a, false, b,
                                                  (short)0, c, false, false);
}

// A fragment 16x32 f16 from a row-major M x K buffer.
// Layout: lane l<16 -> row l, K[0..15];
//         lane l>=16 -> row l-16, K[16..31] (matches frag_b_nt).
static __device__ __forceinline__ v16h frag_a(const f16* __restrict__ tile,
                                              int ldk, int lane) {
    const int row = lane & 15;
    const int c0  = (lane < 16) ? 0 : 16;
    const f16* p  = tile + (size_t)row * ldk + c0;
    v8h lo = *(const v8h*)(p);
    v8h hi = *(const v8h*)(p + 8);
    v16h r;
#pragma unroll
    for (int i = 0; i < 8; ++i) { r[i] = lo[i]; r[i + 8] = hi[i]; }
    return r;
}

// B fragment 32x16 f16 for an NT GEMM: operand stored as N x K row-major.
// Layout used here: lane holds column N = lane%16; lanes 0-15 K[0..15],
// lanes 16-31 K[16..31] -> 16 contiguous halves per lane.
static __device__ __forceinline__ v16h frag_b_nt(const f16* __restrict__ tile,
                                                 int ldk, int lane) {
    const int n  = lane & 15;
    const int k0 = (lane < 16) ? 0 : 16;
    return *(const v16h*)(tile + (size_t)n * ldk + k0);
}

// CDNA5 async copy: 16B global -> LDS, tracked by ASYNCcnt.
static __device__ __forceinline__ void async_ld16(const f16* g, f16* l) {
    asm volatile("global_load_async_to_lds_b128 %0, %1, off"
                 :: "v"((unsigned)(size_t)l),
                    "v"((unsigned long long)(size_t)g)
                 : "memory");
}

// ---------------------------------------------------------------------------
// Kernel 0: f32 -> f16 conversion, 4 elems per thread
// ---------------------------------------------------------------------------
__global__ void cvt_f32_f16_kernel(const float* __restrict__ s,
                                   f16* __restrict__ d, int n) {
    int i = (blockIdx.x * blockDim.x + threadIdx.x) * 4;
    if (i < n) {
        float4 v = *(const float4*)(s + i);
        v4h o = { (f16)v.x, (f16)v.y, (f16)v.z, (f16)v.w };
        *(v4h*)(d + i) = o;
    }
}

// ---------------------------------------------------------------------------
// Kernel 1: QKV GEMM (NT, f16 in / f32 acc) + fused RoPE epilogue.
// Block = 8 waves; wave tile 64(M) x 64(N) = one head slot; block 128 x 256.
// Q is pre-scaled by 1/sqrt(d)*log2(e). V stored transposed [B,NG,HD,T].
// ---------------------------------------------------------------------------
__global__ __launch_bounds__(256)
void qkv_rope_kernel(const f16* __restrict__ xh,      // [B*T, N_EMBD]
                     const f16* __restrict__ wqkv,    // [QKV_DIM, N_EMBD]
                     const float* __restrict__ cosb,  // [T, HD/2]
                     const float* __restrict__ sinb,  // [T, HD/2]
                     f16* __restrict__ qbuf,
                     f16* __restrict__ kbuf,
                     f16* __restrict__ vtbuf) {
    const int lane = threadIdx.x & 31;
    const int wave = threadIdx.x >> 5;
    const int wm   = wave & 1;
    const int wn   = wave >> 1;
    const int m0   = blockIdx.y * 128 + wm * 64;   // flat B*T row base
    const int n0   = blockIdx.x * 256 + wn * 64;   // QKV_DIM col base

    v8f acc[4][4];
#pragma unroll
    for (int i = 0; i < 4; ++i)
#pragma unroll
        for (int j = 0; j < 4; ++j) {
            v8f z = {0.f,0.f,0.f,0.f,0.f,0.f,0.f,0.f};
            acc[i][j] = z;
        }

    for (int k0 = 0; k0 < N_EMBD; k0 += 32) {
        v16h af[4], bf[4];
#pragma unroll
        for (int mt = 0; mt < 4; ++mt)
            af[mt] = frag_a(xh + (size_t)(m0 + mt * 16) * N_EMBD + k0, N_EMBD, lane);
#pragma unroll
        for (int nt = 0; nt < 4; ++nt)
            bf[nt] = frag_b_nt(wqkv + (size_t)(n0 + nt * 16) * N_EMBD + k0, N_EMBD, lane);
#pragma unroll
        for (int mt = 0; mt < 4; ++mt)
#pragma unroll
            for (int nt = 0; nt < 4; ++nt)
                acc[mt][nt] = wmma_f16(af[mt], bf[nt], acc[mt][nt]);
    }

    const int g    = n0 / ((QPK + 2) * HD);
    const int slot = (n0 % ((QPK + 2) * HD)) / HD;
    const int b    = m0 / T_SEQ;
    const int tb   = m0 % T_SEQ;
    const int rhi  = (lane < 16) ? 0 : 8;   // C layout: lanes>=16 hold M+8
    const int nl   = lane & 15;

    if (slot <= QPK) {   // RoPE for q heads (slot 0..3) and k (slot 4)
#pragma unroll
        for (int mt = 0; mt < 4; ++mt) {
#pragma unroll
            for (int nt = 0; nt < 2; ++nt) {
                const int dlo = nt * 16 + nl;          // 0..31
#pragma unroll
                for (int r = 0; r < 8; ++r) {
                    const int t = tb + mt * 16 + rhi + r;
                    const float c  = cosb[(size_t)t * (HD / 2) + dlo];
                    const float s  = sinb[(size_t)t * (HD / 2) + dlo];
                    const float x1 = acc[mt][nt][r];
                    const float x2 = acc[mt][nt + 2][r];
                    acc[mt][nt][r]     = x1 * c - x2 * s;
                    acc[mt][nt + 2][r] = x2 * c + x1 * s;
                }
            }
        }
    }

    if (slot < QPK) {                       // Q head (pre-scaled for softmax)
        const int h = g * QPK + slot;
        f16* dst = qbuf + ((size_t)b * NH + h) * T_SEQ * HD;
#pragma unroll
        for (int mt = 0; mt < 4; ++mt)
#pragma unroll
            for (int nt = 0; nt < 4; ++nt) {
                const int d = nt * 16 + nl;
#pragma unroll
                for (int r = 0; r < 8; ++r) {
                    const int t = tb + mt * 16 + rhi + r;
                    dst[(size_t)t * HD + d] = (f16)(acc[mt][nt][r] * QSCALE);
                }
            }
    } else if (slot == QPK) {               // K
        f16* dst = kbuf + ((size_t)b * NG + g) * T_SEQ * HD;
#pragma unroll
        for (int mt = 0; mt < 4; ++mt)
#pragma unroll
            for (int nt = 0; nt < 4; ++nt) {
                const int d = nt * 16 + nl;
#pragma unroll
                for (int r = 0; r < 8; ++r) {
                    const int t = tb + mt * 16 + rhi + r;
                    dst[(size_t)t * HD + d] = (f16)acc[mt][nt][r];
                }
            }
    } else {                                // V, stored transposed [d][t]
        f16* dst = vtbuf + ((size_t)b * NG + g) * HD * T_SEQ;
#pragma unroll
        for (int mt = 0; mt < 4; ++mt)
#pragma unroll
            for (int nt = 0; nt < 4; ++nt) {
                const int d = nt * 16 + nl;
                const int t = tb + mt * 16 + rhi;   // 8 consecutive t per lane
                v8h pk;
#pragma unroll
                for (int r = 0; r < 8; ++r) pk[r] = (f16)acc[mt][nt][r];
                *(v8h*)(dst + (size_t)d * T_SEQ + t) = pk;  // 16B packed store
            }
    }
}

// ---------------------------------------------------------------------------
// Kernel 2: flash-style attention, one wave = 32 q rows x full head (d=64).
// K/V 64-key chunks staged block-wide in LDS via async-to-LDS (double
// buffered, ASYNCcnt); online softmax in exp2 domain; P goes C-layout ->
// LDS(f16) -> A-fragment layout.
// ---------------------------------------------------------------------------
__global__ __launch_bounds__(256)
void attn_kernel(const f16* __restrict__ qbuf,   // [B,NH,T,HD]
                 const f16* __restrict__ kbuf,   // [B,NG,T,HD]
                 const f16* __restrict__ vtbuf,  // [B,NG,HD,T]
                 f16* __restrict__ obuf) {       // [B*T, N_EMBD]
    __shared__ f16 ktile[2][64 * 64];            // [token][d], 8KB per buffer
    __shared__ f16 vtile[2][64 * 64];            // [d][token], 8KB per buffer
    __shared__ f16 pshare[8][32 * 64];           // per-wave P, 4KB per wave

    const int tid  = threadIdx.x;
    const int lane = tid & 31;
    const int wave = tid >> 5;
    const int bh   = blockIdx.y;                 // b*NH + h
    const int b    = bh >> 5;
    const int h    = bh & 31;
    const int g    = h >> 2;                     // KV group
    const int t0   = blockIdx.x * 256 + wave * 32;

    const f16* qp = qbuf + (((size_t)b * NH + h) * T_SEQ + t0) * HD;
    const f16* kp = kbuf + ((size_t)b * NG + g) * T_SEQ * HD;
    const f16* vp = vtbuf + ((size_t)b * NG + g) * HD * T_SEQ;

    const int nl  = lane & 15;
    const int rhi = (lane < 16) ? 0 : 8;
    const int klo = (lane < 16) ? 0 : 16;

    // Block-wide async prefetch of one 64-key K chunk + V chunk (4 x 16B/thread)
    auto prefetch = [&](int j, int bf) {
#pragma unroll
        for (int c = 0; c < 2; ++c) {
            const int id  = tid + c * 256;       // 0..511
            const int row = id >> 3;             // token
            const int col = (id & 7) * 8;        // half offset in d
            async_ld16(kp + (size_t)(j + row) * HD + col,
                       &ktile[bf][row * 64 + col]);
        }
#pragma unroll
        for (int c = 0; c < 2; ++c) {
            const int id  = tid + c * 256;
            const int row = id >> 3;             // d
            const int col = (id & 7) * 8;        // half offset in token
            async_ld16(vp + (size_t)row * T_SEQ + j + col,
                       &vtile[bf][row * 64 + col]);
        }
    };

    // Preload Q A-fragments (Q already carries 1/sqrt(d)*log2e)
    v16h qf[2][2];
#pragma unroll
    for (int mt = 0; mt < 2; ++mt)
#pragma unroll
        for (int ks = 0; ks < 2; ++ks)
            qf[mt][ks] = frag_a(qp + (size_t)mt * 16 * HD + ks * 32, HD, lane);

    v8f   o[2][4];
    float rmax[2][8], rsum[2][8];
#pragma unroll
    for (int mt = 0; mt < 2; ++mt) {
#pragma unroll
        for (int nt = 0; nt < 4; ++nt) {
            v8f z = {0.f,0.f,0.f,0.f,0.f,0.f,0.f,0.f};
            o[mt][nt] = z;
        }
#pragma unroll
        for (int r = 0; r < 8; ++r) { rmax[mt][r] = -__builtin_inff(); rsum[mt][r] = 0.f; }
    }

    f16* pw = &pshare[wave][0];

    prefetch(0, 0);

    for (int jc = 0; jc < T_SEQ / 64; ++jc) {
        const int bf = jc & 1;
        if (jc + 1 < T_SEQ / 64) {
            prefetch((jc + 1) * 64, bf ^ 1);
            asm volatile("s_wait_asynccnt 4" ::: "memory");  // current chunk done
        } else {
            asm volatile("s_wait_asynccnt 0" ::: "memory");
        }
        __syncthreads();   // all waves' async loads landed in LDS

        // ---- S = Q * K^T from LDS (contract over d=64 = 2 WMMA k-steps) ----
        v8f sacc[2][4];
#pragma unroll
        for (int mt = 0; mt < 2; ++mt)
#pragma unroll
            for (int nt = 0; nt < 4; ++nt) {
                v8f z = {0.f,0.f,0.f,0.f,0.f,0.f,0.f,0.f};
                sacc[mt][nt] = z;
            }
#pragma unroll
        for (int nt = 0; nt < 4; ++nt) {
#pragma unroll
            for (int ks = 0; ks < 2; ++ks) {
                v16h kf = *(const v16h*)&ktile[bf][(nt * 16 + nl) * 64
                                                  + ks * 32 + klo];
#pragma unroll
                for (int mt = 0; mt < 2; ++mt)
                    sacc[mt][nt] = wmma_f16(qf[mt][ks], kf, sacc[mt][nt]);
            }
        }

        // ---- online softmax in exp2 domain (scale folded into Q) ----
#pragma unroll
        for (int mt = 0; mt < 2; ++mt) {
#pragma unroll
            for (int r = 0; r < 8; ++r) {
                float mx = -__builtin_inff();
#pragma unroll
                for (int nt = 0; nt < 4; ++nt)
                    mx = fmaxf(mx, sacc[mt][nt][r]);
#pragma unroll
                for (int off = 8; off >= 1; off >>= 1)
                    mx = fmaxf(mx, __shfl_xor(mx, off, 32));
                const float nm    = fmaxf(rmax[mt][r], mx);
                const float alpha = exp2f(rmax[mt][r] - nm);
                rmax[mt][r] = nm;
                float psum = 0.f;
#pragma unroll
                for (int nt = 0; nt < 4; ++nt) {
                    const float p = exp2f(sacc[mt][nt][r] - nm);
                    sacc[mt][nt][r] = p;
                    psum += p;
                }
#pragma unroll
                for (int off = 8; off >= 1; off >>= 1)
                    psum += __shfl_xor(psum, off, 32);
                rsum[mt][r] = rsum[mt][r] * alpha + psum;
#pragma unroll
                for (int nt = 0; nt < 4; ++nt) o[mt][nt][r] *= alpha;
            }
        }

        // ---- stage P (C layout) to LDS as f16 [m(32)][kc(64)] ----
#pragma unroll
        for (int mt = 0; mt < 2; ++mt)
#pragma unroll
            for (int nt = 0; nt < 4; ++nt) {
                const int kc = nt * 16 + nl;
#pragma unroll
                for (int r = 0; r < 8; ++r) {
                    const int m = mt * 16 + rhi + r;
                    pw[m * 64 + kc] = (f16)sacc[mt][nt][r];
                }
            }
        asm volatile("s_wait_dscnt 0" ::: "memory");  // store->load, wave-private

        // ---- O += P * V from LDS (contract over 64 keys = 2 k-steps) ----
#pragma unroll
        for (int ks = 0; ks < 2; ++ks) {
            v16h pf[2];
#pragma unroll
            for (int mt = 0; mt < 2; ++mt) {
                const f16* pp = pw + (mt * 16 + nl) * 64 + ks * 32
                                + ((lane < 16) ? 0 : 16);
                v8h lo = *(const v8h*)pp;
                v8h hi = *(const v8h*)(pp + 8);
                v16h t;
#pragma unroll
                for (int i = 0; i < 8; ++i) { t[i] = lo[i]; t[i + 8] = hi[i]; }
                pf[mt] = t;
            }
#pragma unroll
            for (int nt = 0; nt < 4; ++nt) {
                v16h vf = *(const v16h*)&vtile[bf][(nt * 16 + nl) * 64
                                                  + ks * 32 + klo];
#pragma unroll
                for (int mt = 0; mt < 2; ++mt)
                    o[mt][nt] = wmma_f16(pf[mt], vf, o[mt][nt]);
            }
        }
        __syncthreads();   // safe to overwrite this buffer next iteration
    }

    // ---- normalize and emit [B*T, N_EMBD] f16 for the projection GEMM ----
    f16* op = obuf + ((size_t)b * T_SEQ + t0) * N_EMBD + h * HD;
#pragma unroll
    for (int mt = 0; mt < 2; ++mt) {
        float rinv[8];
#pragma unroll
        for (int r = 0; r < 8; ++r) rinv[r] = 1.f / rsum[mt][r];
#pragma unroll
        for (int nt = 0; nt < 4; ++nt) {
            const int d = nt * 16 + nl;
#pragma unroll
            for (int r = 0; r < 8; ++r) {
                const int t = mt * 16 + rhi + r;
                op[(size_t)t * N_EMBD + d] = (f16)(o[mt][nt][r] * rinv[r]);
            }
        }
    }
}

// ---------------------------------------------------------------------------
// Kernel 3: output projection GEMM (NT), f32 result to d_out.
// ---------------------------------------------------------------------------
__global__ __launch_bounds__(256)
void proj_kernel(const f16* __restrict__ a,   // [B*T, N_EMBD]
                 const f16* __restrict__ w,   // [N_EMBD, N_EMBD]
                 float* __restrict__ out) {   // [B*T, N_EMBD]
    const int lane = threadIdx.x & 31;
    const int wave = threadIdx.x >> 5;
    const int wm   = wave & 1;
    const int wn   = wave >> 1;
    const int m0   = blockIdx.y * 128 + wm * 64;
    const int n0   = blockIdx.x * 256 + wn * 64;

    v8f acc[4][4];
#pragma unroll
    for (int i = 0; i < 4; ++i)
#pragma unroll
        for (int j = 0; j < 4; ++j) {
            v8f z = {0.f,0.f,0.f,0.f,0.f,0.f,0.f,0.f};
            acc[i][j] = z;
        }

    for (int k0 = 0; k0 < N_EMBD; k0 += 32) {
        v16h af[4], bf[4];
#pragma unroll
        for (int mt = 0; mt < 4; ++mt)
            af[mt] = frag_a(a + (size_t)(m0 + mt * 16) * N_EMBD + k0, N_EMBD, lane);
#pragma unroll
        for (int nt = 0; nt < 4; ++nt)
            bf[nt] = frag_b_nt(w + (size_t)(n0 + nt * 16) * N_EMBD + k0, N_EMBD, lane);
#pragma unroll
        for (int mt = 0; mt < 4; ++mt)
#pragma unroll
            for (int nt = 0; nt < 4; ++nt)
                acc[mt][nt] = wmma_f16(af[mt], bf[nt], acc[mt][nt]);
    }

    const int rhi = (lane < 16) ? 0 : 8;
    const int nl  = lane & 15;
#pragma unroll
    for (int mt = 0; mt < 4; ++mt)
#pragma unroll
        for (int nt = 0; nt < 4; ++nt) {
            const int n = n0 + nt * 16 + nl;
#pragma unroll
            for (int r = 0; r < 8; ++r) {
                const int m = m0 + mt * 16 + rhi + r;
                out[(size_t)m * N_EMBD + n] = acc[mt][nt][r];
            }
        }
}

// ---------------------------------------------------------------------------
// Host launcher
// ---------------------------------------------------------------------------
extern "C" void kernel_launch(void* const* d_in, const int* in_sizes, int n_in,
                              void* d_out, int out_size, void* d_ws, size_t ws_size,
                              hipStream_t stream) {
    const float* x      = (const float*)d_in[0];
    const float* cosb   = (const float*)d_in[1];
    const float* sinb   = (const float*)d_in[2];
    const float* attn_w = (const float*)d_in[3];
    const float* proj_w = (const float*)d_in[4];
    float* out = (float*)d_out;

    // Workspace carve-out (~80 MB), 256B aligned slices.
    char* ws = (char*)d_ws;
    size_t off = 0;
    auto alloc = [&](size_t bytes) {
        char* p = ws + off;
        off += (bytes + 255) & ~(size_t)255;
        return p;
    };
    const size_t MT = (size_t)BATCH * T_SEQ;           // 4096
    f16* xh     = (f16*)alloc(MT * N_EMBD * 2);
    f16* wqkvh  = (f16*)alloc((size_t)QKV_DIM * N_EMBD * 2);
    f16* wprojh = (f16*)alloc((size_t)N_EMBD * N_EMBD * 2);
    f16* qb     = (f16*)alloc((size_t)BATCH * NH * T_SEQ * HD * 2);
    f16* kb     = (f16*)alloc((size_t)BATCH * NG * T_SEQ * HD * 2);
    f16* vtb    = (f16*)alloc((size_t)BATCH * NG * HD * T_SEQ * 2);
    f16* ob     = (f16*)alloc(MT * N_EMBD * 2);
    (void)ws_size; (void)in_sizes; (void)n_in; (void)out_size;

    // 0) f32 -> f16 conversions
    {
        int n1 = (int)(MT * N_EMBD);
        int n2 = QKV_DIM * N_EMBD;
        int n3 = N_EMBD * N_EMBD;
        cvt_f32_f16_kernel<<<(n1 / 4 + 255) / 256, 256, 0, stream>>>(x, xh, n1);
        cvt_f32_f16_kernel<<<(n2 / 4 + 255) / 256, 256, 0, stream>>>(attn_w, wqkvh, n2);
        cvt_f32_f16_kernel<<<(n3 / 4 + 255) / 256, 256, 0, stream>>>(proj_w, wprojh, n3);
    }

    // 1) QKV GEMM + RoPE:  grid (N/256, M/128)
    qkv_rope_kernel<<<dim3(QKV_DIM / 256, (int)(MT / 128)), 256, 0, stream>>>(
        xh, wqkvh, cosb, sinb, qb, kb, vtb);

    // 2) attention: grid (T/256, B*NH)
    attn_kernel<<<dim3(T_SEQ / 256, BATCH * NH), 256, 0, stream>>>(qb, kb, vtb, ob);

    // 3) projection: grid (N/256, M/128)
    proj_kernel<<<dim3(N_EMBD / 256, (int)(MT / 128)), 256, 0, stream>>>(
        ob, wprojh, out);
}
